// FinalHead_27049704030332
// MI455X (gfx1250) — matrix operand order repaired and checked by reading, not compiled, not run on hardware
//
#include <hip/hip_runtime.h>
#include <hip/hip_bf16.h>

#define Bb 8
#define Tt 2048
#define Dd 1024
#define Hh 128

typedef __attribute__((ext_vector_type(16))) __bf16 v16bf;
typedef __attribute__((ext_vector_type(8)))  float  v8f;

union Frag {
    v16bf v;
    uint4 q[2];
    unsigned short u[16];
};

__device__ __forceinline__ unsigned short f32_to_bf16(float f) {
    unsigned int u = __builtin_bit_cast(unsigned int, f);
    u += 0x7FFFu + ((u >> 16) & 1u);   // round-to-nearest-even
    return (unsigned short)(u >> 16);
}

__device__ __forceinline__ v8f wmma_bf16(const Frag& a, const Frag& b, v8f c) {
    return __builtin_amdgcn_wmma_f32_16x16x32_bf16(
        /*neg_a=*/false, a.v, /*neg_b=*/false, b.v,
        /*c_mod=*/(short)0, c, /*reuse_a=*/false, /*reuse_b=*/false);
}

// Async global->LDS copy of 16 bytes per lane (gfx1250, tracked by ASYNCcnt).
__device__ __forceinline__ void async_b128(void* lds_ptr, const void* gptr) {
    unsigned lds_off = (unsigned)(size_t)lds_ptr;   // low 32 bits = LDS byte offset
    asm volatile("global_load_async_to_lds_b128 %0, %1, off"
                 :: "v"(lds_off), "v"(gptr) : "memory");
}

__device__ __forceinline__ void wait_async_8() {
#if __has_builtin(__builtin_amdgcn_s_wait_asynccnt)
    __builtin_amdgcn_s_wait_asynccnt(8);
#else
    asm volatile("s_wait_asynccnt 8" ::: "memory");
#endif
}
__device__ __forceinline__ void wait_async_4() {
#if __has_builtin(__builtin_amdgcn_s_wait_asynccnt)
    __builtin_amdgcn_s_wait_asynccnt(4);
#else
    asm volatile("s_wait_asynccnt 4" ::: "memory");
#endif
}
__device__ __forceinline__ void wait_async_0() {
#if __has_builtin(__builtin_amdgcn_s_wait_asynccnt)
    __builtin_amdgcn_s_wait_asynccnt(0);
#else
    asm volatile("s_wait_asynccnt 0" ::: "memory");
#endif
}

// ---------------------------------------------------------------------------
// Kernel 0: transpose W[D][H] fp32 -> Wt[H][D] bf16 (three weight matrices)
// ---------------------------------------------------------------------------
__global__ __launch_bounds__(256)
void weights_tr_kernel(const float* __restrict__ Wq, const float* __restrict__ Wk,
                       const float* __restrict__ Wv,
                       unsigned short* __restrict__ Tq, unsigned short* __restrict__ Tk,
                       unsigned short* __restrict__ Tv) {
    const int d = blockIdx.x * 256 + threadIdx.x;   // 0..D-1
    const int h = blockIdx.y;                       // 0..H-1
    const float* W = (blockIdx.z == 0) ? Wq : (blockIdx.z == 1) ? Wk : Wv;
    unsigned short* T = (blockIdx.z == 0) ? Tq : (blockIdx.z == 1) ? Tk : Tv;
    T[(size_t)h * Dd + d] = f32_to_bf16(W[(size_t)d * Hh + h]);
}

// ---------------------------------------------------------------------------
// Kernel 1: fused QKV projection. Block = 4 waves x 16 rows = 64 rows x H=128.
// Shared Wt k-tile (32x128 bf16 = 8KB) staged via async->LDS, double buffered.
//   Q,K stored bf16 row-major [B*T][H]; V stored transposed bf16 [B][H][T].
// ---------------------------------------------------------------------------
__global__ __launch_bounds__(128, 4)   // cap 256 VGPRs: no vgpr-msb churn
void qkv_proj_kernel(const float* __restrict__ x, const float* __restrict__ att,
                     const float* __restrict__ bk, const float* __restrict__ bq,
                     const float* __restrict__ bv,
                     const unsigned short* __restrict__ Wtq,
                     const unsigned short* __restrict__ Wtk,
                     const unsigned short* __restrict__ Wtv,
                     unsigned short* __restrict__ Qbf,
                     unsigned short* __restrict__ Kbf,
                     unsigned short* __restrict__ Vt) {
    // wtile[buf] layout: [h=128][k=32] bf16 (8KB per buffer)
    __shared__ __attribute__((aligned(16))) unsigned short wtile[2][4096];

    const int tid  = threadIdx.x;
    const int lane = tid & 31;
    const int wave = tid >> 5;
    const int r0   = blockIdx.x * 64 + wave * 16;
    const int m    = lane & 15;
    const int hi   = lane >> 4;

    for (int proj = 0; proj < 3; ++proj) {
        const float* A = (proj == 2) ? x : att;
        const unsigned short* Wt = (proj == 0) ? Wtq : (proj == 1) ? Wtk : Wtv;
        const float* bias = (proj == 0) ? bq : (proj == 1) ? bk : bv;

        v8f c[8];
        #pragma unroll
        for (int nt = 0; nt < 8; ++nt) c[nt] = v8f{0, 0, 0, 0, 0, 0, 0, 0};

        // prologue: issue async copy of Wt k-tile 0 into buffer 0
        #pragma unroll
        for (int cc = 0; cc < 4; ++cc) {
            const int cch = tid + cc * 128;            // chunk 0..511 (16B each)
            const int h = cch >> 2, koff = (cch & 3) * 8;
            async_b128(&wtile[0][cch * 8], Wt + (size_t)h * Dd + koff);
        }

        const float* arow = A + (size_t)(r0 + m) * Dd;
        for (int kt = 0; kt < Dd / 32; ++kt) {
            const int k0  = kt * 32;
            const int buf = kt & 1;
            if (kt + 1 < Dd / 32) {
                #pragma unroll
                for (int cc = 0; cc < 4; ++cc) {
                    const int cch = tid + cc * 128;
                    const int h = cch >> 2, koff = (cch & 3) * 8;
                    async_b128(&wtile[buf ^ 1][cch * 8],
                               Wt + (size_t)h * Dd + (k0 + 32) + koff);
                }
                wait_async_4();
            } else {
                wait_async_0();
            }
            __syncthreads();   // tile kt resident in wtile[buf]

            if (k0 + 32 < Dd)
                __builtin_prefetch(arow + k0 + 32, 0, 1);

            // A fragment: 16 fp32 -> bf16 (two contiguous chunks of 8)
            Frag af;
            float4 f0 = *(const float4*)(arow + k0 + hi * 8);
            float4 f1 = *(const float4*)(arow + k0 + hi * 8 + 4);
            float4 f2 = *(const float4*)(arow + k0 + 16 + hi * 8);
            float4 f3 = *(const float4*)(arow + k0 + 16 + hi * 8 + 4);
            af.u[0]  = f32_to_bf16(f0.x); af.u[1]  = f32_to_bf16(f0.y);
            af.u[2]  = f32_to_bf16(f0.z); af.u[3]  = f32_to_bf16(f0.w);
            af.u[4]  = f32_to_bf16(f1.x); af.u[5]  = f32_to_bf16(f1.y);
            af.u[6]  = f32_to_bf16(f1.z); af.u[7]  = f32_to_bf16(f1.w);
            af.u[8]  = f32_to_bf16(f2.x); af.u[9]  = f32_to_bf16(f2.y);
            af.u[10] = f32_to_bf16(f2.z); af.u[11] = f32_to_bf16(f2.w);
            af.u[12] = f32_to_bf16(f3.x); af.u[13] = f32_to_bf16(f3.y);
            af.u[14] = f32_to_bf16(f3.z); af.u[15] = f32_to_bf16(f3.w);

            #pragma unroll
            for (int nt = 0; nt < 8; ++nt) {
                // B fragment: column n = m -> wtile row (nt*16+m), K contiguous
                Frag bf;
                const unsigned short* wr =
                    &wtile[buf][(nt * 16 + m) * 32 + hi * 8];
                bf.q[0] = ((const uint4*)wr)[0];
                bf.q[1] = ((const uint4*)wr)[2];
                c[nt] = wmma_bf16(af, bf, c[nt]);
            }
            __syncthreads();   // done reading wtile[buf] before it is rewritten
        }

        // epilogue: add bias, convert, store
        #pragma unroll
        for (int nt = 0; nt < 8; ++nt) {
            const int col = nt * 16 + m;
            const float bval = bias[col];
            #pragma unroll
            for (int j = 0; j < 8; ++j) {
                const int row = r0 + j + 8 * hi;
                unsigned short h16 = f32_to_bf16(c[nt][j] + bval);
                if (proj == 0)      Qbf[(size_t)row * Hh + col] = h16;
                else if (proj == 1) Kbf[(size_t)row * Hh + col] = h16;
                else {
                    const int bidx = row >> 11;          // row / T
                    const int t    = row & (Tt - 1);     // row % T
                    Vt[(size_t)bidx * Hh * Tt + (size_t)col * Tt + t] = h16;
                }
            }
        }
    }
}

// ---------------------------------------------------------------------------
// Kernel 2: flash attention. Block = 4 waves x 16 query rows = 64 rows.
// Shared K tile (32x128) and V tile (128x32) staged via async->LDS, double
// buffered; P re-layout via LDS. Softmax in exp2 domain (v_exp_f32 is base-2).
// ---------------------------------------------------------------------------
__global__ __launch_bounds__(128, 4)   // cap 256 VGPRs: no vgpr-msb churn
void flash_attn_kernel(const unsigned short* __restrict__ Qbf,
                       const unsigned short* __restrict__ Kbf,
                       const unsigned short* __restrict__ Vt,
                       float* __restrict__ out) {
    __shared__ __attribute__((aligned(16))) unsigned short ktile[2][4096]; // [key32][h128]
    __shared__ __attribute__((aligned(16))) unsigned short vtile[2][4096]; // [h128][key32]
    __shared__ __attribute__((aligned(16))) unsigned short plds[4][16][32];

    const int tid  = threadIdx.x;
    const int lane = tid & 31;
    const int wave = tid >> 5;
    const int b    = blockIdx.y;
    const int q0   = blockIdx.x * 64 + wave * 16;
    const int m    = lane & 15;
    const int hi   = lane >> 4;
    // sqrt(128) (faithful reference scaling) folded with log2(e): exp2 domain
    const float scale2 = 11.313708498984761f * 1.4426950408889634f;

    const unsigned short* Kbase = Kbf + (size_t)b * Tt * Hh;
    const unsigned short* Vbase = Vt + (size_t)b * Hh * Tt;

    // Preload Q fragments for all 4 K-steps over H
    Frag qf[4];
    {
        const unsigned short* qrow = Qbf + ((size_t)b * Tt + q0 + m) * Hh;
        #pragma unroll
        for (int kk = 0; kk < 4; ++kk) {
            qf[kk].q[0] = *(const uint4*)(qrow + kk * 32 + hi * 16);
            qf[kk].q[1] = *(const uint4*)(qrow + kk * 32 + hi * 16 + 8);
        }
    }

    v8f o[8];
    float rm[8], rl[8];
    #pragma unroll
    for (int ht = 0; ht < 8; ++ht) o[ht] = v8f{0, 0, 0, 0, 0, 0, 0, 0};
    #pragma unroll
    for (int j = 0; j < 8; ++j) { rm[j] = -1.0e30f; rl[j] = 0.0f; }

    // issue tile copies: K tile is one contiguous 8KB block; V tile strided rows
    auto issue_tile = [&](int buf, int tk0) {
        const unsigned short* kg = Kbase + (size_t)tk0 * Hh;
        #pragma unroll
        for (int cc = 0; cc < 4; ++cc) {
            const int c = tid + cc * 128;               // 16B chunks
            async_b128(&ktile[buf][c * 8], kg + c * 8);
        }
        #pragma unroll
        for (int cc = 0; cc < 4; ++cc) {
            const int c = tid + cc * 128;
            const int h = c >> 2, toff = (c & 3) * 8;
            async_b128(&vtile[buf][c * 8], Vbase + (size_t)h * Tt + tk0 + toff);
        }
    };

    issue_tile(0, 0);   // prologue

    for (int kt = 0; kt < Tt / 32; ++kt) {
        const int tk0 = kt * 32;
        const int buf = kt & 1;
        if (kt + 1 < Tt / 32) {
            issue_tile(buf ^ 1, tk0 + 32);
            wait_async_8();
        } else {
            wait_async_0();
        }
        __syncthreads();   // tile kt resident

        // ---- S = Q K^T over two 16-column groups, operands from LDS ----
        v8f s0 = v8f{0, 0, 0, 0, 0, 0, 0, 0};
        v8f s1 = v8f{0, 0, 0, 0, 0, 0, 0, 0};
        #pragma unroll
        for (int kk = 0; kk < 4; ++kk) {
            Frag kf;
            const unsigned short* k0p = &ktile[buf][(m)*128 + kk * 32 + hi * 16];
            kf.q[0] = ((const uint4*)k0p)[0];
            kf.q[1] = ((const uint4*)k0p)[1];
            s0 = wmma_bf16(qf[kk], kf, s0);
            const unsigned short* k1p = &ktile[buf][(16 + m) * 128 + kk * 32 + hi * 16];
            kf.q[0] = ((const uint4*)k1p)[0];
            kf.q[1] = ((const uint4*)k1p)[1];
            s1 = wmma_bf16(qf[kk], kf, s1);
        }

        // ---- online softmax per row (row j of this half), exp2 domain ----
        #pragma unroll
        for (int j = 0; j < 8; ++j) {
            float a0 = s0[j] * scale2;
            float a1 = s1[j] * scale2;
            float mx = fmaxf(a0, a1);
            #pragma unroll
            for (int msk = 1; msk < 16; msk <<= 1)
                mx = fmaxf(mx, __shfl_xor(mx, msk, 32));
            const float mnew = fmaxf(rm[j], mx);
            const float corr = exp2f(rm[j] - mnew);
            rm[j] = mnew;
            const float p0 = exp2f(a0 - mnew);
            const float p1 = exp2f(a1 - mnew);
            plds[wave][j + 8 * hi][m]      = f32_to_bf16(p0);
            plds[wave][j + 8 * hi][16 + m] = f32_to_bf16(p1);
            float rs = p0 + p1;
            #pragma unroll
            for (int msk = 1; msk < 16; msk <<= 1)
                rs += __shfl_xor(rs, msk, 32);
            rl[j] = rl[j] * corr + rs;
            #pragma unroll
            for (int ht = 0; ht < 8; ++ht) o[ht][j] *= corr;
        }

        __syncthreads();   // P ready; ktile reads complete

        // ---- O += P V, operands from LDS ----
        Frag pf;
        pf.q[0] = *(const uint4*)&plds[wave][m][hi * 16];
        pf.q[1] = *(const uint4*)&plds[wave][m][hi * 16 + 8];
        #pragma unroll
        for (int ht = 0; ht < 8; ++ht) {
            Frag vf;
            const unsigned short* vr = &vtile[buf][(ht * 16 + m) * 32 + hi * 16];
            vf.q[0] = ((const uint4*)vr)[0];
            vf.q[1] = ((const uint4*)vr)[1];
            o[ht] = wmma_bf16(pf, vf, o[ht]);
        }
        __syncthreads();   // all reads of buffers done before overwrite
    }

    // ---- epilogue: normalize and store fp32 ----
    #pragma unroll
    for (int ht = 0; ht < 8; ++ht) {
        const int col = ht * 16 + m;
        #pragma unroll
        for (int j = 0; j < 8; ++j) {
            const int row = q0 + j + 8 * hi;
            out[((size_t)b * Tt + row) * Hh + col] = o[ht][j] / rl[j];
        }
    }
}

// ---------------------------------------------------------------------------
extern "C" void kernel_launch(void* const* d_in, const int* in_sizes, int n_in,
                              void* d_out, int out_size, void* d_ws, size_t ws_size,
                              hipStream_t stream) {
    const float* x   = (const float*)d_in[0];
    const float* att = (const float*)d_in[1];
    const float* Wk  = (const float*)d_in[2];
    const float* bk  = (const float*)d_in[3];
    const float* Wq  = (const float*)d_in[4];
    const float* bq  = (const float*)d_in[5];
    const float* Wv  = (const float*)d_in[6];
    const float* bv  = (const float*)d_in[7];
    float* out = (float*)d_out;

    char* ws = (char*)d_ws;
    const size_t wt_bytes  = (size_t)Dd * Hh * 2;        // 256 KB each
    const size_t qkv_bytes = (size_t)Bb * Tt * Hh * 2;   // 4 MB each
    unsigned short* Wtq = (unsigned short*)(ws);
    unsigned short* Wtk = (unsigned short*)(ws + wt_bytes);
    unsigned short* Wtv = (unsigned short*)(ws + 2 * wt_bytes);
    unsigned short* Qbf = (unsigned short*)(ws + 3 * wt_bytes);
    unsigned short* Kbf = (unsigned short*)(ws + 3 * wt_bytes + qkv_bytes);
    unsigned short* Vt  = (unsigned short*)(ws + 3 * wt_bytes + 2 * qkv_bytes);

    weights_tr_kernel<<<dim3(Dd / 256, Hh, 3), 256, 0, stream>>>(
        Wq, Wk, Wv, Wtq, Wtk, Wtv);
    qkv_proj_kernel<<<dim3((Bb * Tt) / 64), 128, 0, stream>>>(
        x, att, bk, bq, bv, Wtq, Wtk, Wtv, Qbf, Kbf, Vt);
    flash_attn_kernel<<<dim3(Tt / 64, Bb), 128, 0, stream>>>(Qbf, Kbf, Vt, out);
}
